// MyDecoder_76768245448759
// MI455X (gfx1250) — matrix operand order repaired and checked by reading, not compiled, not run on hardware
//
#include <hip/hip_runtime.h>
#include <math.h>

typedef _Float16 f16;
typedef __attribute__((ext_vector_type(4)))  _Float16 v4h;
typedef __attribute__((ext_vector_type(8)))  _Float16 v8h;
typedef __attribute__((ext_vector_type(16))) _Float16 v16h;
typedef __attribute__((ext_vector_type(8)))  float    v8f;

#define E_DIM  1024
#define NHEAD  16
#define HDIM   64
#define NLAYER 4
#define BATCH  2
#define SEQ    1024
#define TSEQ   1024
#define FFDIM  4096
#define LN_EPS 1e-5f

static __device__ __forceinline__ v8f wmma_f16(v16h a, v16h b, v8f c) {
  return __builtin_amdgcn_wmma_f32_16x16x32_f16(false, a, false, b,
                                                (short)0, c, false, false);
}
static __device__ __forceinline__ v8h ld8(const f16* p) { return *(const v8h*)p; }
static __device__ __forceinline__ v16h cat8(v8h a, v8h b) {
  return __builtin_shufflevector(a, b, 0,1,2,3,4,5,6,7,8,9,10,11,12,13,14,15);
}
static __device__ __forceinline__ unsigned lds_off(const void* p) {
  return (unsigned)(unsigned long long)p;   // generic LDS addr: low 32 bits = LDS offset
}
// CDNA5 async global->LDS copy (16B per lane), tracked by ASYNCcnt.
static __device__ __forceinline__ void async_ld_b128(unsigned ldsaddr, const void* gaddr) {
  asm volatile("global_load_async_to_lds_b128 %0, %1, off"
               :: "v"(ldsaddr), "v"(gaddr) : "memory");
}
static __device__ __forceinline__ void wait_async0() {
  asm volatile("s_wait_asynccnt 0" ::: "memory");
}
static __device__ __forceinline__ void wait_async4() {
  asm volatile("s_wait_asynccnt 4" ::: "memory");
}

// ---------------------------------------------------------------------------
// fp32 -> f16 elementwise convert (4 elems/thread)
// ---------------------------------------------------------------------------
__global__ __launch_bounds__(256)
void cvt_h_kernel(const float* __restrict__ in, f16* __restrict__ out)
{
  size_t i = ((size_t)blockIdx.x * 256 + threadIdx.x) * 4;
  float4 a = *(const float4*)(in + i);
  v4h h = { (f16)a.x, (f16)a.y, (f16)a.z, (f16)a.w };
  *(v4h*)(out + i) = h;
}

// ---------------------------------------------------------------------------
// Transpose + fp32->f16 convert: out[c*R + r] = (f16)in[r*C + c]
// ---------------------------------------------------------------------------
__global__ __launch_bounds__(256)
void transpose_h_kernel(const float* __restrict__ in, f16* __restrict__ out,
                        int R, int C)
{
  __shared__ f16 Ts[32][33];
  const int r0 = blockIdx.y * 32, c0 = blockIdx.x * 32;
  const int tr = threadIdx.x >> 5;
  const int tc = threadIdx.x & 31;
#pragma unroll
  for (int j = 0; j < 4; ++j)
    Ts[tr + 8 * j][tc] = (f16)in[(size_t)(r0 + tr + 8 * j) * C + (c0 + tc)];
  __syncthreads();
#pragma unroll
  for (int j = 0; j < 4; ++j)
    out[(size_t)(c0 + tr + 8 * j) * R + (r0 + tc)] = Ts[tc][tr + 8 * j];
}

// ---------------------------------------------------------------------------
// GEMM: C[M,N] = act(A[M,K] @ W[K,N] + bias[N]).
// A: f16 [M][K]; Wt: f16 pre-transposed [N][K]. Block tile 128x128, K-step 32.
// Both tiles staged via async global->LDS with double buffering: issue tile
// k+1, s_wait_asynccnt 4 (tile k's 4 ops retired in order), barrier, 8 WMMAs.
// ---------------------------------------------------------------------------
template <int ACT_GELU, int OUT_F16>
__global__ __launch_bounds__(256)
void gemm_kernel(const f16* __restrict__ A, const f16* __restrict__ Wt,
                 const float* __restrict__ bias,
                 float* __restrict__ Cf, f16* __restrict__ Ch,
                 int M, int N, int K)
{
  __shared__ f16 As[2][128][40];   // [buf][m][k], row 80B
  __shared__ f16 Bs[2][128][40];   // [buf][n][k]

  const int t    = threadIdx.x;
  const int lane = t & 31;
  const int wave = t >> 5;
  const int wm   = wave & 1;
  const int wn   = wave >> 1;
  const int hlf  = lane >> 4;
  const int l16  = lane & 15;
  const int m0   = blockIdx.y * 128;
  const int n0   = blockIdx.x * 128;

  const int rch = t >> 2, cch = (t & 3) * 8;      // 16B chunk: row, k-offset
  const int rc2 = (t + 256) >> 2;                 // second chunk row

  v8f acc[4][2] = {};

  const int nk = K >> 5;
  auto issue = [&](int it, int buf) {
    const int k0 = it << 5;
    async_ld_b128(lds_off(&As[buf][rch][cch]), A  + (size_t)(m0 + rch) * K + k0 + cch);
    async_ld_b128(lds_off(&Bs[buf][rch][cch]), Wt + (size_t)(n0 + rch) * K + k0 + cch);
    async_ld_b128(lds_off(&As[buf][rc2][cch]), A  + (size_t)(m0 + rc2) * K + k0 + cch);
    async_ld_b128(lds_off(&Bs[buf][rc2][cch]), Wt + (size_t)(n0 + rc2) * K + k0 + cch);
  };

  issue(0, 0);
  for (int it = 0; it < nk; ++it) {
    const int buf = it & 1;
    if (it + 1 < nk) { issue(it + 1, buf ^ 1); wait_async4(); }
    else             { wait_async0(); }
    __syncthreads();

    v16h af[4], bf[2];
#pragma unroll
    for (int fm = 0; fm < 4; ++fm) {
      int row = wm * 64 + fm * 16 + l16;
      af[fm] = cat8(ld8(&As[buf][row][hlf * 16]), ld8(&As[buf][row][hlf * 16 + 8]));
    }
#pragma unroll
    for (int fn = 0; fn < 2; ++fn) {
      int col = wn * 32 + fn * 16 + l16;
      bf[fn] = cat8(ld8(&Bs[buf][col][hlf * 16]), ld8(&Bs[buf][col][hlf * 16 + 8]));
    }
#pragma unroll
    for (int fm = 0; fm < 4; ++fm)
#pragma unroll
      for (int fn = 0; fn < 2; ++fn)
        acc[fm][fn] = wmma_f16(af[fm], bf[fn], acc[fm][fn]);
    __syncthreads();   // done reading buf; next iteration may overwrite it
  }

#pragma unroll
  for (int fm = 0; fm < 4; ++fm)
#pragma unroll
    for (int fn = 0; fn < 2; ++fn)
#pragma unroll
      for (int v = 0; v < 8; ++v) {
        int row = m0 + wm * 64 + fm * 16 + hlf * 8 + v;
        int col = n0 + wn * 32 + fn * 16 + l16;
        float x = acc[fm][fn][v] + bias[col];
        if (ACT_GELU) x = 0.5f * x * (1.0f + erff(x * 0.70710678118654752f));
        if (OUT_F16) Ch[(size_t)row * N + col] = (f16)x;
        else         Cf[(size_t)row * N + col] = x;
      }
}

// ---------------------------------------------------------------------------
// Flash attention: 128 threads = 4 waves; one 16-query tile per wave; block
// shares one 32-key K/V LDS tile (K async). Output f16 (feeds O-proj GEMM).
// ---------------------------------------------------------------------------
__global__ __launch_bounds__(128)
void attn_kernel(const f16* __restrict__ Q, const f16* __restrict__ Km,
                 const f16* __restrict__ V, f16* __restrict__ O,
                 int Sq, int Sk, int causal)
{
  __shared__ f16 Ks[32][72];       // [key][d]   row 144B
  __shared__ f16 Vt[64][40];       // [d][key]   row 80B
  __shared__ f16 Pst[4][16][40];   // per-wave P, row 80B

  const int t    = threadIdx.x;
  const int lane = t & 31;
  const int wave = t >> 5;
  const int hlf  = lane >> 4;
  const int l16  = lane & 15;
  const int h    = blockIdx.y;
  const int b    = blockIdx.z;
  const int q0   = (blockIdx.x * 4 + wave) * 16;
  const int q0mx = (blockIdx.x * 4 + 3) * 16;

  const size_t qbase  = ((size_t)b * Sq + q0) * E_DIM + (size_t)h * HDIM;
  const size_t kvbase = (size_t)b * Sk * E_DIM + (size_t)h * HDIM;

  v16h qf[2];
#pragma unroll
  for (int c = 0; c < 2; ++c) {
    const f16* qp = Q + qbase + (size_t)l16 * E_DIM + c * 32;
    qf[c] = cat8(ld8(qp + hlf * 16), ld8(qp + hlf * 16 + 8));
  }

  v8f o[4] = {};
  float mrow[8], lrow[8];
#pragma unroll
  for (int v = 0; v < 8; ++v) { mrow[v] = -INFINITY; lrow[v] = 0.0f; }

  const int nkb = causal ? ((q0mx + 15) / 32 + 1) : (Sk / 32);
  const float scale = 0.125f;  // 1/sqrt(64)

  for (int kt = 0; kt < nkb; ++kt) {
    const int key0 = kt * 32;
    __syncthreads();
#pragma unroll
    for (int j = 0; j < 2; ++j) {
      int ch = t + 128 * j;
      int key = ch >> 3, seg = ch & 7;
      async_ld_b128(lds_off(&Ks[key][seg * 8]),
                    Km + kvbase + (size_t)(key0 + key) * E_DIM + seg * 8);
    }
    {
      int key = t & 31, d0 = (t >> 5) * 16;
      const f16* vp = V + kvbase + (size_t)(key0 + key) * E_DIM + d0;
      v8h va = ld8(vp), vb = ld8(vp + 8);
#pragma unroll
      for (int j = 0; j < 8; ++j) Vt[d0 + j][key]     = va[j];
#pragma unroll
      for (int j = 0; j < 8; ++j) Vt[d0 + 8 + j][key] = vb[j];
    }
    wait_async0();
    __syncthreads();

    const bool active = (!causal) || (key0 <= q0 + 15);
    if (active) {
      v8f sa[2];
#pragma unroll
      for (int s = 0; s < 2; ++s) {
        v8f sc = {};
#pragma unroll
        for (int c = 0; c < 2; ++c) {
          const f16* kp = &Ks[s * 16 + l16][c * 32 + hlf * 16];
          v16h kf = cat8(ld8(kp), ld8(kp + 8));
          sc = wmma_f16(qf[c], kf, sc);
        }
        sa[s] = sc;
      }
#pragma unroll
      for (int s = 0; s < 2; ++s)
#pragma unroll
        for (int v = 0; v < 8; ++v) {
          float x = sa[s][v] * scale;
          int key  = key0 + s * 16 + l16;
          int qrow = q0 + hlf * 8 + v;
          if (causal && key > qrow) x += -1e9f;
          sa[s][v] = x;
        }
#pragma unroll
      for (int v = 0; v < 8; ++v) {
        float mx = fmaxf(sa[0][v], sa[1][v]);
#pragma unroll
        for (int off = 8; off >= 1; off >>= 1)
          mx = fmaxf(mx, __shfl_xor(mx, off, 16));
        float mn    = fmaxf(mrow[v], mx);
        float alpha = __expf(mrow[v] - mn);
        float p0    = __expf(sa[0][v] - mn);
        float p1    = __expf(sa[1][v] - mn);
        float rs    = p0 + p1;
#pragma unroll
        for (int off = 8; off >= 1; off >>= 1)
          rs += __shfl_xor(rs, off, 16);
        lrow[v] = lrow[v] * alpha + rs;
        mrow[v] = mn;
#pragma unroll
        for (int f = 0; f < 4; ++f) o[f][v] *= alpha;
        int row = hlf * 8 + v;
        Pst[wave][row][l16]      = (f16)p0;
        Pst[wave][row][16 + l16] = (f16)p1;
      }
      v16h pf = cat8(ld8(&Pst[wave][l16][hlf * 16]),
                     ld8(&Pst[wave][l16][hlf * 16 + 8]));
#pragma unroll
      for (int f = 0; f < 4; ++f) {
        const f16* vp = &Vt[f * 16 + l16][hlf * 16];
        v16h vf = cat8(ld8(vp), ld8(vp + 8));
        o[f] = wmma_f16(pf, vf, o[f]);
      }
    }
  }

#pragma unroll
  for (int f = 0; f < 4; ++f)
#pragma unroll
    for (int v = 0; v < 8; ++v) {
      int row = hlf * 8 + v;
      O[((size_t)b * Sq + q0 + row) * E_DIM + (size_t)h * HDIM + f * 16 + l16] =
          (f16)(o[f][v] / lrow[v]);
    }
}

// ---------------------------------------------------------------------------
// Residual + LayerNorm, dual output: fp32 (residual chain) + optional f16
// (feeds the next GEMMs). In-place over xin is safe.
// ---------------------------------------------------------------------------
__global__ __launch_bounds__(256)
void ln_kernel(const float* __restrict__ xin, const float* __restrict__ hres,
               const float* __restrict__ g, const float* __restrict__ bt,
               float* __restrict__ out, f16* __restrict__ outh)
{
  __shared__ float red[256];
  const int t = threadIdx.x;
  const size_t base = (size_t)blockIdx.x * E_DIM;

  float x[4];
  float s = 0.0f;
#pragma unroll
  for (int j = 0; j < 4; ++j) {
    x[j] = xin[base + t + 256 * j] + hres[base + t + 256 * j];
    s += x[j];
  }
  red[t] = s;
  __syncthreads();
  for (int o2 = 128; o2 > 0; o2 >>= 1) {
    if (t < o2) red[t] += red[t + o2];
    __syncthreads();
  }
  float mean = red[0] / (float)E_DIM;
  __syncthreads();

  float vs = 0.0f;
#pragma unroll
  for (int j = 0; j < 4; ++j) { float d = x[j] - mean; vs += d * d; }
  red[t] = vs;
  __syncthreads();
  for (int o2 = 128; o2 > 0; o2 >>= 1) {
    if (t < o2) red[t] += red[t + o2];
    __syncthreads();
  }
  float rstd = rsqrtf(red[0] / (float)E_DIM + LN_EPS);
  __syncthreads();

#pragma unroll
  for (int j = 0; j < 4; ++j) {
    int col = t + 256 * j;
    float y = (x[j] - mean) * rstd * g[col] + bt[col];
    out[base + col] = y;
    if (outh) outh[base + col] = (f16)y;
  }
}

// ---------------------------------------------------------------------------
extern "C" void kernel_launch(void* const* d_in, const int* in_sizes, int n_in,
                              void* d_out, int out_size, void* d_ws, size_t ws_size,
                              hipStream_t stream) {
  (void)in_sizes; (void)n_in; (void)out_size; (void)ws_size;

  const float* x_in = (const float*)d_in[0];
  const float* enc  = (const float*)d_in[1];
  // d_in[2] = mask (causal tril; applied analytically)
  const float* sw  = (const float*)d_in[3];
  const float* sb  = (const float*)d_in[4];
  const float* cw  = (const float*)d_in[5];
  const float* cb  = (const float*)d_in[6];
  const float* f1w = (const float*)d_in[7];
  const float* f1b = (const float*)d_in[8];
  const float* f2w = (const float*)d_in[9];
  const float* f2b = (const float*)d_in[10];
  const float* lng = (const float*)d_in[11];
  const float* lnb = (const float*)d_in[12];
  float* out = (float*)d_out;

  char* ws = (char*)d_ws;
  size_t off = 0;
  auto carve = [&](size_t bytes) -> void* {
    off = (off + 255) & ~(size_t)255;
    void* p = ws + off;
    off += bytes;
    return p;
  };

  const int M = BATCH * SEQ;                 // 2048 rows (== B*T)
  const size_t EE = (size_t)E_DIM * E_DIM;
  const size_t EF = (size_t)E_DIM * FFDIM;
  const size_t ME = (size_t)M * E_DIM;

  float* X    = (float*)carve(ME * sizeof(float));   // fp32 residual chain
  float* P1   = (float*)carve(ME * sizeof(float));   // sublayer output fp32
  f16*   Xh   = (f16*)carve(ME * sizeof(f16));       // f16 copy of X
  f16*   X0h  = (f16*)carve(ME * sizeof(f16));       // f16 copy of x input
  f16*   Ench = (f16*)carve(ME * sizeof(f16));       // f16 copy of encoder out
  f16*   AOh  = (f16*)carve(ME * sizeof(f16));       // attention out f16
  f16*   Hffh = (f16*)carve((size_t)M * FFDIM * sizeof(f16));
  f16*   Qh   = (f16*)carve(ME * sizeof(f16));
  f16*   Kh   = (f16*)carve(ME * sizeof(f16));
  f16*   Vh   = (f16*)carve(ME * sizeof(f16));
  f16*   WT   = (f16*)carve((8 * EE + 2 * EF) * sizeof(f16));

  f16* swT = WT;                 // 4 x [E][E]
  f16* cwT = WT + 4 * EE;        // 4 x [E][E]
  f16* f1T = WT + 8 * EE;        // [FF][E]
  f16* f2T = f1T + EF;           // [E][FF]

  const dim3 blk(256);
  const dim3 gE(E_DIM / 128, M / 128);
  const dim3 gF(FFDIM / 128, M / 128);
  const dim3 gA(SEQ / 64, NHEAD, BATCH);
  const dim3 gTE(E_DIM / 32, E_DIM / 32);
  const dim3 gT1(FFDIM / 32, E_DIM / 32);
  const dim3 gT2(E_DIM / 32, FFDIM / 32);

  // one-time f16 copies of the two fp32 activations we read as GEMM inputs
  cvt_h_kernel<<<dim3(ME / 1024), blk, 0, stream>>>(x_in, X0h);
  cvt_h_kernel<<<dim3(ME / 1024), blk, 0, stream>>>(enc,  Ench);

  const float* cur  = x_in;   // fp32 residual input
  const f16*   curh = X0h;    // f16 GEMM input
  for (int l = 0; l < NLAYER; ++l) {
    // ---- weight transpose + f16 convert for this layer ----
    for (int j = 0; j < 4; ++j) {
      transpose_h_kernel<<<gTE, blk, 0, stream>>>(sw + ((size_t)l * 4 + j) * EE,
                                                  swT + (size_t)j * EE, E_DIM, E_DIM);
      transpose_h_kernel<<<gTE, blk, 0, stream>>>(cw + ((size_t)l * 4 + j) * EE,
                                                  cwT + (size_t)j * EE, E_DIM, E_DIM);
    }
    transpose_h_kernel<<<gT1, blk, 0, stream>>>(f1w + (size_t)l * EF, f1T, E_DIM, FFDIM);
    transpose_h_kernel<<<gT2, blk, 0, stream>>>(f2w + (size_t)l * EF, f2T, FFDIM, E_DIM);

    const float* bb  = sb + (size_t)l * 4 * E_DIM;
    const float* bb2 = cb + (size_t)l * 4 * E_DIM;

    // ---- masked self-attention ----
    gemm_kernel<0, 1><<<gE, blk, 0, stream>>>(curh, swT + 0 * EE, bb + 0 * E_DIM,
                                              nullptr, Qh, M, E_DIM, E_DIM);
    gemm_kernel<0, 1><<<gE, blk, 0, stream>>>(curh, swT + 1 * EE, bb + 1 * E_DIM,
                                              nullptr, Kh, M, E_DIM, E_DIM);
    gemm_kernel<0, 1><<<gE, blk, 0, stream>>>(curh, swT + 2 * EE, bb + 2 * E_DIM,
                                              nullptr, Vh, M, E_DIM, E_DIM);
    attn_kernel<<<gA, dim3(128), 0, stream>>>(Qh, Kh, Vh, AOh, SEQ, SEQ, 1);
    gemm_kernel<0, 0><<<gE, blk, 0, stream>>>(AOh, swT + 3 * EE, bb + 3 * E_DIM,
                                              P1, nullptr, M, E_DIM, E_DIM);
    ln_kernel<<<dim3(M), blk, 0, stream>>>(cur, P1, lng + ((size_t)l * 3 + 0) * E_DIM,
                                           lnb + ((size_t)l * 3 + 0) * E_DIM, X, Xh);

    // ---- encoder-decoder cross-attention ----
    gemm_kernel<0, 1><<<gE, blk, 0, stream>>>(Xh, cwT + 0 * EE, bb2 + 0 * E_DIM,
                                              nullptr, Qh, M, E_DIM, E_DIM);
    gemm_kernel<0, 1><<<gE, blk, 0, stream>>>(Ench, cwT + 1 * EE, bb2 + 1 * E_DIM,
                                              nullptr, Kh, M, E_DIM, E_DIM);
    gemm_kernel<0, 1><<<gE, blk, 0, stream>>>(Ench, cwT + 2 * EE, bb2 + 2 * E_DIM,
                                              nullptr, Vh, M, E_DIM, E_DIM);
    attn_kernel<<<gA, dim3(128), 0, stream>>>(Qh, Kh, Vh, AOh, SEQ, TSEQ, 0);
    gemm_kernel<0, 0><<<gE, blk, 0, stream>>>(AOh, cwT + 3 * EE, bb2 + 3 * E_DIM,
                                              P1, nullptr, M, E_DIM, E_DIM);
    ln_kernel<<<dim3(M), blk, 0, stream>>>(X, P1, lng + ((size_t)l * 3 + 1) * E_DIM,
                                           lnb + ((size_t)l * 3 + 1) * E_DIM, X, Xh);

    // ---- FFN (exact GELU) ----
    gemm_kernel<1, 1><<<gF, blk, 0, stream>>>(Xh, f1T, f1b + (size_t)l * FFDIM,
                                              nullptr, Hffh, M, FFDIM, E_DIM);
    gemm_kernel<0, 0><<<gE, blk, 0, stream>>>(Hffh, f2T, f2b + (size_t)l * E_DIM,
                                              P1, nullptr, M, E_DIM, FFDIM);
    float* dst = (l == NLAYER - 1) ? out : X;
    f16*  dsth = (l == NLAYER - 1) ? (f16*)nullptr : Xh;
    ln_kernel<<<dim3(M), blk, 0, stream>>>(X, P1, lng + ((size_t)l * 3 + 2) * E_DIM,
                                           lnb + ((size_t)l * 3 + 2) * E_DIM, dst, dsth);
    cur  = X;
    curh = Xh;
  }
}
